// TimeLSTMHyp_41472204210323
// MI455X (gfx1250) — matrix operand order repaired and checked by reading, not compiled, not run on hardware
//
#include <hip/hip_runtime.h>
#include <hip/hip_bf16.h>
#include <math.h>

// ---------------------------------------------------------------------------
// Hyperbolic Time-LSTM for MI455X (gfx1250, wave32, WMMA bf16 16x16x32,
// TDM tensor_load_to_lds for A-panel staging).
// ---------------------------------------------------------------------------

#define B_   64
#define S_   128
#define D_   768
#define H_   768
#define H4_  3072
#define KDIM 768          // every GEMM in this model has K = 768
#define MIN_NORM 1e-7f

typedef __bf16 bf16_t;
typedef __attribute__((ext_vector_type(16))) __bf16 v16bf;
typedef __attribute__((ext_vector_type(8)))  __bf16 v8bf;
typedef __attribute__((ext_vector_type(8)))  float  v8f;
typedef __attribute__((ext_vector_type(4)))  unsigned int u32x4;
typedef __attribute__((ext_vector_type(8)))  int i32x8;
typedef __attribute__((ext_vector_type(4)))  int i32x4;

// ---------------------------------------------------------------------------
// math helpers (curvature k = c, sq = sqrt(-k))
// ---------------------------------------------------------------------------
__device__ inline float dev_artan_k(float x, float sq) {
    float v = sq * x;
    v = fminf(fmaxf(v, -1.0f + 1e-6f), 1.0f - 1e-6f);
    return atanhf(v) / sq;
}
__device__ inline float dev_tan_k(float x, float sq) {
    return tanhf(sq * x) / sq;
}
__device__ inline float dev_mob_add(float x, float y, float xy, float x2, float y2, float k) {
    float num = (1.0f - 2.0f * k * xy - k * y2) * x + (1.0f + k * x2) * y;
    float den = 1.0f - 2.0f * k * xy + k * k * x2 * y2;
    return num / fmaxf(den, MIN_NORM);
}
__device__ inline float dev_sigmoid(float x) { return 1.0f / (1.0f + expf(-x)); }

// multi-value block reduction; blockDim.x == 256 (8 wave32's). sred: >= 80 floats
template <int NR>
__device__ inline void block_reduce(float* v, float* sred) {
    const int lane = threadIdx.x & 31;
    const int wave = threadIdx.x >> 5;
#pragma unroll
    for (int off = 16; off > 0; off >>= 1)
#pragma unroll
        for (int q = 0; q < NR; ++q)
            v[q] += __shfl_down(v[q], off, 32);
    if (lane == 0)
#pragma unroll
        for (int q = 0; q < NR; ++q) sred[wave * NR + q] = v[q];
    __syncthreads();
    if (threadIdx.x < NR) {
        float acc = 0.0f;
        for (int w = 0; w < 8; ++w) acc += sred[w * NR + threadIdx.x];
        sred[64 + threadIdx.x] = acc;
    }
    __syncthreads();
#pragma unroll
    for (int q = 0; q < NR; ++q) v[q] = sred[64 + q];
    __syncthreads();
}

// ---------------------------------------------------------------------------
// conversions
// ---------------------------------------------------------------------------
__global__ void f32_to_bf16_kernel(const float* __restrict__ src, bf16_t* __restrict__ dst, int n) {
    int i = blockIdx.x * blockDim.x + threadIdx.x;
    if (i < n) dst[i] = (bf16_t)src[i];
}

// WB[c][k] = W_all[c % H][(c / H) * H + k]   (per-gate HxH transpose), c in [0,4H), k in [0,H)
__global__ void convert_Wall_kernel(const float* __restrict__ W_all, bf16_t* __restrict__ WB) {
    int i = blockIdx.x * blockDim.x + threadIdx.x;
    if (i >= H4_ * H_) return;
    int c = i / H_;
    int kk = i - c * H_;
    WB[i] = (bf16_t)W_all[(size_t)(c % H_) * H4_ + (c / H_) * H_ + kk];
}

__global__ void init_state_kernel(const float* __restrict__ h0, const float* __restrict__ c0,
                                  float* __restrict__ hF, float* __restrict__ cF,
                                  bf16_t* __restrict__ hB, bf16_t* __restrict__ cB, int n) {
    int i = blockIdx.x * blockDim.x + threadIdx.x;
    if (i < n) {
        float h = h0[i], c = c0[i];
        hF[i] = h; cF[i] = c;
        hB[i] = (bf16_t)h; cB[i] = (bf16_t)c;
    }
}

// ---------------------------------------------------------------------------
// bf16 WMMA GEMM:  C[M,N] = A[M,K] * B(N,K)^T,  K == 768 fixed.
// block = 256 threads = 8 waves; block tile 16(M) x 256(N);
// each wave owns two 16x16 output subtiles (reuses its A fragment).
// A panel (16x768 bf16 = 24 KB) staged once into LDS by the Tensor Data Mover.
// ---------------------------------------------------------------------------
__global__ void gemm_bf16_wmma_kernel(const bf16_t* __restrict__ A, const bf16_t* __restrict__ Bw,
                                      float* __restrict__ C, int M, int N) {
    __shared__ __attribute__((aligned(16))) bf16_t As[16 * KDIM];

    const int lane = threadIdx.x & 31;
    const int wave = threadIdx.x >> 5;
    const int half = lane >> 4;   // 0 or 1
    const int mn   = lane & 15;
    const int m0   = blockIdx.y * 16;
    const int n0   = blockIdx.x * 256 + wave * 32;

    // ---- stage A panel: rows m0..m0+15, all 768 K, via TDM (wave 0 only) ----
    if (wave == 0) {
        unsigned long long gaddr =
            (unsigned long long)(const void*)(A + (size_t)m0 * KDIM);
        unsigned int lds_addr = (unsigned int)(size_t)(void*)&As[0];

        // D# group 0: count=1 | lds_addr | global_addr(57b) | type=2
        u32x4 g0;
        g0[0] = 1u;                                   // count=1, user descriptor
        g0[1] = lds_addr;                             // LDS byte address
        g0[2] = (unsigned int)gaddr;                  // global_addr[31:0]
        g0[3] = ((unsigned int)(gaddr >> 32) & 0x01FFFFFFu) | (2u << 30); // [56:32] | type=2

        // D# group 1: data_size=1 (2B), tensor 768x16, tile 768x16, stride 768
        i32x8 g1;
        g1[0] = (int)(1u << 16);                      // data_size=1 (2 bytes)
        g1[1] = (int)((KDIM & 0xFFFF) << 16);         // tensor_dim0[15:0] @ bits 63:48
        g1[2] = (int)((16u & 0xFFFF) << 16);          // tensor_dim0[31:16]=0 | tensor_dim1[15:0]=16
        g1[3] = (int)((KDIM & 0xFFFFu) << 16);        // tensor_dim1[31:16]=0 | tile_dim0=768
        g1[4] = (int)16;                              // tile_dim1=16 | tile_dim2=0
        g1[5] = (int)KDIM;                            // tensor_dim0_stride[31:0] = 768
        g1[6] = 0;                                    // stride hi | tensor_dim1_stride lo
        g1[7] = 0;                                    // tensor_dim1_stride hi

        i32x4 gz4 = {0, 0, 0, 0};                     // groups 2/3: tile_dim2/3/4 = 0 (unused)
        i32x8 gz8 = {0, 0, 0, 0, 0, 0, 0, 0};

        __builtin_amdgcn_tensor_load_to_lds(g0, g1, gz4, gz4, gz8, 0);
        __builtin_amdgcn_s_wait_tensorcnt(0);
    }
    __syncthreads();

    v8f acc0 = {};
    v8f acc1 = {};

    for (int k0 = 0; k0 < KDIM; k0 += 32) {
        // A fragment (contiguous K order, matching B):
        //  lanes 0-15 : e0-15 -> K 0-15   (M = lane)
        //  lanes16-31 : e0-15 -> K 16-31  (M = lane-16)
        v8bf alo = *(const v8bf*)&As[mn * KDIM + k0 + half * 16];
        v8bf ahi = *(const v8bf*)&As[mn * KDIM + k0 + half * 16 + 8];
        v16bf afrag = __builtin_shufflevector(alo, ahi,
            0,1,2,3,4,5,6,7,8,9,10,11,12,13,14,15);

        // B fragments: lane holds column n = mn, K = half*16 + e (contiguous in (N,K))
        const bf16_t* bp0 = Bw + (size_t)(n0 + mn) * KDIM + k0 + half * 16;
        const bf16_t* bp1 = Bw + (size_t)(n0 + 16 + mn) * KDIM + k0 + half * 16;
        v8bf b0lo = *(const v8bf*)bp0;
        v8bf b0hi = *(const v8bf*)(bp0 + 8);
        v8bf b1lo = *(const v8bf*)bp1;
        v8bf b1hi = *(const v8bf*)(bp1 + 8);
        v16bf bfrag0 = __builtin_shufflevector(b0lo, b0hi,
            0,1,2,3,4,5,6,7,8,9,10,11,12,13,14,15);
        v16bf bfrag1 = __builtin_shufflevector(b1lo, b1hi,
            0,1,2,3,4,5,6,7,8,9,10,11,12,13,14,15);

        acc0 = __builtin_amdgcn_wmma_f32_16x16x32_bf16(
            false, afrag, false, bfrag0, (short)0, acc0, false, false);
        acc1 = __builtin_amdgcn_wmma_f32_16x16x32_bf16(
            false, afrag, false, bfrag1, (short)0, acc1, false, false);
    }

    // C/D layout: VGPR r -> M = half*8 + r, N = mn
#pragma unroll
    for (int rr = 0; rr < 8; ++rr) {
        int m = m0 + half * 8 + rr;
        C[(size_t)m * N + n0 + mn]      = acc0[rr];
        C[(size_t)m * N + n0 + 16 + mn] = acc1[rr];
    }
}

// ---------------------------------------------------------------------------
// phase-1 rowwise transform: M2[r][g] = mobius_matvec(U_g, x_r) given raw Ux.
// one block per (r, g); also stores ||m2||^2.
// ---------------------------------------------------------------------------
__global__ void ux_transform_kernel(const float* __restrict__ X, float* __restrict__ M2,
                                    float* __restrict__ m2n2, const float* __restrict__ kp) {
    __shared__ float sred[80];
    const int r = blockIdx.x;
    const int g = blockIdx.y;
    const float k = *kp;
    const float sq = sqrtf(-k);

    float xv[3], mv[3];
    float v[2] = {0.0f, 0.0f};
#pragma unroll
    for (int q = 0; q < 3; ++q) {
        int j = threadIdx.x + q * 256;
        xv[q] = X[(size_t)r * D_ + j];
        mv[q] = M2[(size_t)r * H4_ + g * H_ + j];
        v[0] += xv[q] * xv[q];
        v[1] += mv[q] * mv[q];
    }
    block_reduce<2>(v, sred);
    float xn  = fmaxf(sqrtf(v[0]), MIN_NORM);
    float mxn = fmaxf(sqrtf(v[1]), MIN_NORM);
    float scale = dev_tan_k(mxn / xn * dev_artan_k(xn, sq), sq) / mxn;

    float y2 = 0.0f;
#pragma unroll
    for (int q = 0; q < 3; ++q) {
        int j = threadIdx.x + q * 256;
        float m = mv[q] * scale;
        M2[(size_t)r * H4_ + g * H_ + j] = m;
        y2 += m * m;
    }
    float vy[1] = {y2};
    block_reduce<1>(vy, sred);
    if (threadIdx.x == 0) m2n2[r * 4 + g] = vy[0];
}

// ---------------------------------------------------------------------------
// per-step kernel 1: cell-path hyperbolic math -> c_adj. one block per b.
// ---------------------------------------------------------------------------
__global__ void step1_kernel(const float* __restrict__ cF, const float* __restrict__ Wdc,
                             const float* __restrict__ T, int s,
                             float* __restrict__ c_adj, const float* __restrict__ kp) {
    __shared__ float sred[80];
    const int b = blockIdx.x;
    const float k = *kp;
    const float sq = sqrtf(-k);
    const float t = T[b * S_ + s];

    float cc[3], wd[3];
#pragma unroll
    for (int q = 0; q < 3; ++q) {
        int j = threadIdx.x + q * 256;
        cc[q] = cF[(size_t)b * H_ + j];
        wd[q] = Wdc[(size_t)b * H_ + j];
    }

    // ||cc||, ||Wd*cc||
    float v2[2] = {0, 0};
#pragma unroll
    for (int q = 0; q < 3; ++q) { v2[0] += cc[q] * cc[q]; v2[1] += wd[q] * wd[q]; }
    block_reduce<2>(v2, sred);
    float ncc2 = v2[0];
    float ncc = fmaxf(sqrtf(v2[0]), MIN_NORM);
    float nwd = fmaxf(sqrtf(v2[1]), MIN_NORM);

    // mv = mobius_matvec(W_d, cc) = mvs * wd
    float mvs = dev_tan_k(nwd / ncc * dev_artan_k(ncc, sq), sq) / nwd;

    float v1[1] = {0};
#pragma unroll
    for (int q = 0; q < 3; ++q) { float m = mvs * wd[q]; v1[0] += m * m; }
    block_reduce<1>(v1, sred);
    float nmv = fmaxf(sqrtf(v1[0]), MIN_NORM);
    float ls = dev_artan_k(nmv, sq) / nmv;   // logmap0 scale

    // th = tanh(logmap0(mv)) elementwise
    float th[3];
    v1[0] = 0;
#pragma unroll
    for (int q = 0; q < 3; ++q) { th[q] = tanhf(ls * mvs * wd[q]); v1[0] += th[q] * th[q]; }
    block_reduce<1>(v1, sred);
    float nth = fmaxf(sqrtf(v1[0]), MIN_NORM);
    float es = dev_tan_k(nth, sq) / nth;     // expmap0 scale

    float s1[3];
#pragma unroll
    for (int q = 0; q < 3; ++q) s1[q] = es * th[q];

    // c_s2 = mobius_pmul(c_s1, t_exp): w = c_s1, x = t_exp
    float v3[3] = {0, 0, 0};
#pragma unroll
    for (int q = 0; q < 3; ++q) {
        float wx = s1[q] * t;
        v3[0] += wx * wx;
        v3[1] += s1[q] * s1[q];
        v3[2] += -s1[q] * cc[q];
    }
    block_reduce<3>(v3, sred);
    float xn  = fmaxf(sqrtf((float)H_ * t * t), MIN_NORM);   // ||t_exp||
    float wxn = fmaxf(sqrtf(v3[0]), MIN_NORM);
    float ps  = dev_tan_k(wxn / xn * dev_artan_k(xn, sq), sq) / wxn;

    float s2[3], cl[3];
#pragma unroll
    for (int q = 0; q < 3; ++q) {
        s2[q] = ps * (s1[q] * t);
        cl[q] = dev_mob_add(-s1[q], cc[q], v3[2], v3[1], ncc2, k);
    }

    // c_adj = mobius_add(c_l, c_s2)
    float v3b[3] = {0, 0, 0};
#pragma unroll
    for (int q = 0; q < 3; ++q) {
        v3b[0] += cl[q] * s2[q];
        v3b[1] += cl[q] * cl[q];
        v3b[2] += s2[q] * s2[q];
    }
    block_reduce<3>(v3b, sred);
#pragma unroll
    for (int q = 0; q < 3; ++q) {
        int j = threadIdx.x + q * 256;
        c_adj[(size_t)b * H_ + j] = dev_mob_add(cl[q], s2[q], v3b[0], v3b[1], v3b[2], k);
    }
}

// ---------------------------------------------------------------------------
// per-step kernel 2: gates + state update + output. one block per b.
// ---------------------------------------------------------------------------
__global__ void step2_kernel(float* __restrict__ hF, float* __restrict__ cF,
                             const float* __restrict__ Wh, const float* __restrict__ M2,
                             const float* __restrict__ m2n2, const float* __restrict__ c_adj,
                             int s, float* __restrict__ out,
                             bf16_t* __restrict__ hB, bf16_t* __restrict__ cB,
                             const float* __restrict__ kp) {
    __shared__ float sred[80];
    const int b = blockIdx.x;
    const int r = b * S_ + s;
    const float k = *kp;
    const float sq = sqrtf(-k);

    float h[3], wh[4][3], m2v[4][3], cadj[3];
#pragma unroll
    for (int q = 0; q < 3; ++q) {
        int j = threadIdx.x + q * 256;
        h[q]    = hF[(size_t)b * H_ + j];
        cadj[q] = c_adj[(size_t)b * H_ + j];
#pragma unroll
        for (int g = 0; g < 4; ++g) {
            wh[g][q]  = Wh[(size_t)b * H4_ + g * H_ + j];
            m2v[g][q] = M2[(size_t)r * H4_ + g * H_ + j];
        }
    }

    // ||h||^2 + per-gate ||Wh||^2
    float v5[5] = {0, 0, 0, 0, 0};
#pragma unroll
    for (int q = 0; q < 3; ++q) {
        v5[0] += h[q] * h[q];
#pragma unroll
        for (int g = 0; g < 4; ++g) v5[1 + g] += wh[g][q] * wh[g][q];
    }
    block_reduce<5>(v5, sred);
    float nh = fmaxf(sqrtf(v5[0]), MIN_NORM);
    float ath = dev_artan_k(nh, sq);

    float m1s[4], x2g[4];
#pragma unroll
    for (int g = 0; g < 4; ++g) {
        float nwh = fmaxf(sqrtf(v5[1 + g]), MIN_NORM);
        m1s[g] = dev_tan_k(nwh / nh * ath, sq) / nwh;
        x2g[g] = m1s[g] * m1s[g] * v5[1 + g];   // ||m1||^2
    }

    // <m1_g, m2_g>
    float v4[4] = {0, 0, 0, 0};
#pragma unroll
    for (int q = 0; q < 3; ++q)
#pragma unroll
        for (int g = 0; g < 4; ++g) v4[g] += (m1s[g] * wh[g][q]) * m2v[g][q];
    block_reduce<4>(v4, sred);

    // pre_g = mobius_add(m1_g, m2_g); then ||pre||^2 per gate
    float pre[4][3];
    float v4b[4] = {0, 0, 0, 0};
#pragma unroll
    for (int q = 0; q < 3; ++q)
#pragma unroll
        for (int g = 0; g < 4; ++g) {
            float p = dev_mob_add(m1s[g] * wh[g][q], m2v[g][q],
                                  v4[g], x2g[g], m2n2[r * 4 + g], k);
            pre[g][q] = p;
            v4b[g] += p * p;
        }
    block_reduce<4>(v4b, sred);

    // gates = sigmoid(logmap0(pre)); order: 0=f, 1=i, 2=o, 3=c_tmp
    float gate[4][3];
#pragma unroll
    for (int g = 0; g < 4; ++g) {
        float np = fmaxf(sqrtf(v4b[g]), MIN_NORM);
        float lg = dev_artan_k(np, sq) / np;
#pragma unroll
        for (int q = 0; q < 3; ++q) gate[g][q] = dev_sigmoid(lg * pre[g][q]);
    }

    // p1 = mobius_pmul(i, c_tmp), p2 = mobius_pmul(f, c_adj)
    float vp[4] = {0, 0, 0, 0};
#pragma unroll
    for (int q = 0; q < 3; ++q) {
        float ct = gate[3][q];
        float ic = gate[1][q] * ct;
        float fc = gate[0][q] * cadj[q];
        vp[0] += ct * ct;
        vp[1] += ic * ic;
        vp[2] += cadj[q] * cadj[q];
        vp[3] += fc * fc;
    }
    block_reduce<4>(vp, sred);
    float xn1 = fmaxf(sqrtf(vp[0]), MIN_NORM), wxn1 = fmaxf(sqrtf(vp[1]), MIN_NORM);
    float xn2 = fmaxf(sqrtf(vp[2]), MIN_NORM), wxn2 = fmaxf(sqrtf(vp[3]), MIN_NORM);
    float p1s = dev_tan_k(wxn1 / xn1 * dev_artan_k(xn1, sq), sq) / wxn1;
    float p2s = dev_tan_k(wxn2 / xn2 * dev_artan_k(xn2, sq), sq) / wxn2;

    float p1[3], p2[3];
    float v3[3] = {0, 0, 0};
#pragma unroll
    for (int q = 0; q < 3; ++q) {
        p1[q] = p1s * (gate[1][q] * gate[3][q]);
        p2[q] = p2s * (gate[0][q] * cadj[q]);
        v3[0] += p1[q] * p2[q];
        v3[1] += p1[q] * p1[q];
        v3[2] += p2[q] * p2[q];
    }
    block_reduce<3>(v3, sred);

    // new_c = mobius_add(p1, p2)
    float nc[3], th[3];
    float v1[1] = {0};
#pragma unroll
    for (int q = 0; q < 3; ++q) {
        nc[q] = dev_mob_add(p1[q], p2[q], v3[0], v3[1], v3[2], k);
        th[q] = tanhf(nc[q]);
        v1[0] += th[q] * th[q];
    }
    block_reduce<1>(v1, sred);
    float nthn = fmaxf(sqrtf(v1[0]), MIN_NORM);
    float esc = dev_tan_k(nthn, sq) / nthn;   // expmap0 scale

    // new_h = mobius_pmul(o, expmap0(tanh(new_c)))
    float e[3];
    float ve[2] = {0, 0};
#pragma unroll
    for (int q = 0; q < 3; ++q) {
        e[q] = esc * th[q];
        float oe = gate[2][q] * e[q];
        ve[0] += e[q] * e[q];
        ve[1] += oe * oe;
    }
    block_reduce<2>(ve, sred);
    float xne = fmaxf(sqrtf(ve[0]), MIN_NORM), wxne = fmaxf(sqrtf(ve[1]), MIN_NORM);
    float hs = dev_tan_k(wxne / xne * dev_artan_k(xne, sq), sq) / wxne;

#pragma unroll
    for (int q = 0; q < 3; ++q) {
        int j = threadIdx.x + q * 256;
        float newh = hs * (gate[2][q] * e[q]);
        hF[(size_t)b * H_ + j] = newh;
        cF[(size_t)b * H_ + j] = nc[q];
        hB[(size_t)b * H_ + j] = (bf16_t)newh;
        cB[(size_t)b * H_ + j] = (bf16_t)nc[q];
        out[(size_t)r * H_ + j] = gate[2][q];   // outputs = o
    }
}

// ---------------------------------------------------------------------------
// host orchestration
// ---------------------------------------------------------------------------
extern "C" void kernel_launch(void* const* d_in, const int* in_sizes, int n_in,
                              void* d_out, int out_size, void* d_ws, size_t ws_size,
                              hipStream_t stream) {
    (void)in_sizes; (void)n_in; (void)out_size; (void)ws_size;

    const float* inputs = (const float*)d_in[0];   // (B,S,D)
    const float* tstamp = (const float*)d_in[1];   // (B,S)
    const float* h0     = (const float*)d_in[2];   // (B,H)
    const float* c0     = (const float*)d_in[3];   // (B,H)
    const float* W_all  = (const float*)d_in[4];   // (H,4H)
    const float* U_all  = (const float*)d_in[5];   // (4H,D)
    const float* W_d    = (const float*)d_in[6];   // (H,H)
    const float* kp     = (const float*)d_in[7];   // scalar

    char* ws = (char*)d_ws;
    size_t off = 0;
    auto alloc = [&](size_t bytes) -> char* {
        char* p = ws + off;
        off = (off + bytes + 255) & ~(size_t)255;
        return p;
    };
    bf16_t* Xb   = (bf16_t*)alloc((size_t)B_ * S_ * D_ * 2);
    bf16_t* UB   = (bf16_t*)alloc((size_t)H4_ * D_ * 2);
    bf16_t* WB   = (bf16_t*)alloc((size_t)H4_ * H_ * 2);
    bf16_t* WD   = (bf16_t*)alloc((size_t)H_ * H_ * 2);
    float*  M2   = (float*) alloc((size_t)B_ * S_ * H4_ * 4);
    float*  m2n2 = (float*) alloc((size_t)B_ * S_ * 4 * 4);
    float*  hF   = (float*) alloc((size_t)B_ * H_ * 4);
    float*  cF   = (float*) alloc((size_t)B_ * H_ * 4);
    bf16_t* hB   = (bf16_t*)alloc((size_t)B_ * H_ * 2);
    bf16_t* cB   = (bf16_t*)alloc((size_t)B_ * H_ * 2);
    float*  Wh   = (float*) alloc((size_t)B_ * H4_ * 4);
    float*  Wdc  = (float*) alloc((size_t)B_ * H_ * 4);
    float*  cadj = (float*) alloc((size_t)B_ * H_ * 4);

    float* out = (float*)d_out;

    // ---- conversions ----
    {
        int n = B_ * S_ * D_;
        f32_to_bf16_kernel<<<(n + 255) / 256, 256, 0, stream>>>(inputs, Xb, n);
    }
    {
        int n = H4_ * D_;
        f32_to_bf16_kernel<<<(n + 255) / 256, 256, 0, stream>>>(U_all, UB, n);
    }
    {
        int n = H_ * H_;
        f32_to_bf16_kernel<<<(n + 255) / 256, 256, 0, stream>>>(W_d, WD, n);
    }
    {
        int n = H4_ * H_;
        convert_Wall_kernel<<<(n + 255) / 256, 256, 0, stream>>>(W_all, WB);
    }

    // ---- phase 1: raw Ux GEMM (8192x768 @ 768x3072) + rowwise mobius transform ----
    {
        dim3 grid(H4_ / 256, (B_ * S_) / 16);
        gemm_bf16_wmma_kernel<<<grid, 256, 0, stream>>>(Xb, UB, M2, B_ * S_, H4_);
    }
    {
        dim3 grid(B_ * S_, 4);
        ux_transform_kernel<<<grid, 256, 0, stream>>>(inputs, M2, m2n2, kp);
    }

    // ---- init recurrent state ----
    {
        int n = B_ * H_;
        init_state_kernel<<<(n + 255) / 256, 256, 0, stream>>>(h0, c0, hF, cF, hB, cB, n);
    }

    // ---- sequential scan over S steps ----
    for (int s = 0; s < S_; ++s) {
        {   // Wh = h @ W^T  (64 x 3072, K=768)
            dim3 grid(H4_ / 256, B_ / 16);
            gemm_bf16_wmma_kernel<<<grid, 256, 0, stream>>>(hB, WB, Wh, B_, H4_);
        }
        {   // Wdc = cc @ W_d^T (64 x 768, K=768)
            dim3 grid(H_ / 256, B_ / 16);
            gemm_bf16_wmma_kernel<<<grid, 256, 0, stream>>>(cB, WD, Wdc, B_, H_);
        }
        step1_kernel<<<B_, 256, 0, stream>>>(cF, Wdc, tstamp, s, cadj, kp);
        step2_kernel<<<B_, 256, 0, stream>>>(hF, cF, Wh, M2, m2n2, cadj, s, out, hB, cB, kp);
    }

    // ---- h_last, c_last into d_out tail ----
    (void)hipMemcpyAsync(out + (size_t)B_ * S_ * H_, hF,
                         (size_t)B_ * H_ * sizeof(float), hipMemcpyDeviceToDevice, stream);
    (void)hipMemcpyAsync(out + (size_t)B_ * S_ * H_ + (size_t)B_ * H_, cF,
                         (size_t)B_ * H_ * sizeof(float), hipMemcpyDeviceToDevice, stream);
}
